// Attention_88648124991143
// MI455X (gfx1250) — matrix operand order repaired and checked by reading, not compiled, not run on hardware
//
#include <hip/hip_runtime.h>

// ---------------- problem constants ----------------
constexpr int B    = 2;
constexpr int S    = 2048;
constexpr int D    = 4096;
constexpr int NH   = 32;
constexpr int NKV  = 8;
constexpr int HD   = D / NH;          // 128
constexpr int NREP = NH / NKV;        // 4
constexpr int HD2  = HD / 2;          // 64

typedef unsigned short u16;
typedef unsigned int   u32;
typedef __attribute__((ext_vector_type(16))) __bf16 v16bf;
typedef __attribute__((ext_vector_type(8)))  float  v8f;

struct __align__(16) B128 { u32 w[4]; };
union Frag16 {
    v16bf bf;
    B128  q[2];
    u16   h[16];
};

__device__ __forceinline__ u16 f32_to_bf16(float f) {
    u32 u = __float_as_uint(f);
    u32 r = u + 0x7FFFu + ((u >> 16) & 1u);   // round to nearest even
    return (u16)(r >> 16);
}
__device__ __forceinline__ float bf16_to_f32(u16 h) {
    return __uint_as_float(((u32)h) << 16);
}

// ---------- CDNA5 async global->LDS copy (guarded; fallback always compiles) ----------
#ifndef __has_builtin
#define __has_builtin(x) 0
#endif
#if __has_builtin(__builtin_amdgcn_global_load_async_to_lds_b128)
#define HAVE_ASYNC_LDS 1
#else
#define HAVE_ASYNC_LDS 0
#endif

#define AS1 __attribute__((address_space(1)))
#define AS3 __attribute__((address_space(3)))
typedef int v4i_vs __attribute__((vector_size(16)));   // matches builtin param pointee

__device__ __forceinline__ void copy16_g2l(const u16* g, u16* l) {
#if HAVE_ASYNC_LDS
    __builtin_amdgcn_global_load_async_to_lds_b128(
        (AS1 v4i_vs*)(unsigned long long)g,
        (AS3 v4i_vs*)(unsigned int)(unsigned long long)l, 0, 0);
#else
    *(B128*)l = *(const B128*)g;        // global_load_b128 + ds_store_b128
#endif
}

__device__ __forceinline__ void async_join() {
#if HAVE_ASYNC_LDS
#if __has_builtin(__builtin_amdgcn_s_wait_asynccnt)
    __builtin_amdgcn_s_wait_asynccnt(0);
#else
    asm volatile("s_wait_asynccnt 0" ::: "memory");
#endif
#endif
}

// ---------------- elementwise cast fp32 -> bf16 ----------------
__global__ void cast_bf16_kernel(const float* __restrict__ in, u16* __restrict__ out, int n) {
    int i = blockIdx.x * blockDim.x + threadIdx.x;
    if (i < n) out[i] = f32_to_bf16(in[i]);
}

// ---------------- cast + transpose: W (N x K fp32, row-major) -> Wt (K x N bf16) ----
__global__ void transpose_cast_kernel(const float* __restrict__ W, u16* __restrict__ Wt,
                                      int N, int K) {
    __shared__ float tile[32][33];
    const int kb = blockIdx.x * 32;
    const int nb = blockIdx.y * 32;
    const int tx = threadIdx.x;   // 0..31
    const int ty = threadIdx.y;   // 0..7
    #pragma unroll
    for (int i = ty; i < 32; i += 8)
        tile[i][tx] = W[(size_t)(nb + i) * K + kb + tx];
    __syncthreads();
    #pragma unroll
    for (int i = ty; i < 32; i += 8)
        Wt[(size_t)(kb + i) * N + nb + tx] = f32_to_bf16(tile[tx][i]);
}

// ---------------- LDS-staged double-buffered bf16 WMMA GEMM ----------------
//  C(MxN) = A(MxK) @ Bt(KxN).  Block tile 128x128, k-step 32, 256 threads = 8 waves.
//  Waves arranged 4(M) x 2(N); wave tile 32x64 = 2x4 WMMA tiles (8 wmma / k-step).
//  LDS rows padded (A: 40 u16, B: 136 u16) for conflict-free 16-lane ds_load_b128.
constexpr int GA_STRIDE = 40;    // 32 + 8 pad (80 B rows)
constexpr int GB_STRIDE = 136;   // 128 + 8 pad (272 B rows)

template <bool OUT_BF16>
__global__ __launch_bounds__(256)
void gemm_bf16_kernel(const u16* __restrict__ A, const u16* __restrict__ Bt,
                      void* __restrict__ Cout, int M, int N, int K) {
    __shared__ __align__(16) u16 Al[2][128 * GA_STRIDE];
    __shared__ __align__(16) u16 Bl[2][32 * GB_STRIDE];

    const int tid  = threadIdx.x;
    const int lane = tid & 31;
    const int w    = tid >> 5;
    const int hi   = lane >> 4;
    const int lo   = lane & 15;
    const int bm   = blockIdx.y * 128;
    const int bn   = blockIdx.x * 128;
    const int wm   = (w >> 1) * 32;       // 0,32,64,96
    const int wn   = (w & 1) * 64;        // 0,64

    // cooperative copy assignment (one b128 each for A and B per thread per stage)
    const int arow = tid >> 1, aseg = (tid & 1) * 16;   // A tile: 128 rows x 32 cols
    const int brow = tid >> 3, bseg = (tid & 7) * 16;   // B tile:  32 rows x 128 cols

    auto stage_load = [&](int s, int k) {
        copy16_g2l(A  + (size_t)(bm + arow) * K + k + aseg,
                   &Al[s][arow * GA_STRIDE + aseg]);
        copy16_g2l(Bt + (size_t)(k + brow) * N + bn + bseg,
                   &Bl[s][brow * GB_STRIDE + bseg]);
    };

    v8f acc[2][4] = {};
    stage_load(0, 0);
    const int nk = K / 32;
    for (int kt = 0; kt < nk; ++kt) {
        const int s = kt & 1;
        async_join();          // my async stores to stage s done
        __syncthreads();       // everyone's stage-s data visible; stage s^1 free
        if (kt + 1 < nk) stage_load(s ^ 1, (kt + 1) * 32);

        Frag16 af[2], bfr[4];
        #pragma unroll
        for (int i = 0; i < 2; ++i) {
            const u16* ap = &Al[s][(wm + 16 * i + lo) * GA_STRIDE + hi * 8];
            af[i].q[0] = *(const B128*)ap;
            af[i].q[1] = *(const B128*)(ap + 16);
        }
        #pragma unroll
        for (int j = 0; j < 4; ++j) {
            const u16* bp = &Bl[s][wn + 16 * j + lo];
            #pragma unroll
            for (int e = 0; e < 16; ++e)
                bfr[j].h[e] = bp[((e < 8) ? (hi * 8 + e) : (16 + hi * 8 + e - 8)) * GB_STRIDE];
        }
        #pragma unroll
        for (int i = 0; i < 2; ++i)
            #pragma unroll
            for (int j = 0; j < 4; ++j)
                acc[i][j] = __builtin_amdgcn_wmma_f32_16x16x32_bf16(
                    false, af[i].bf, false, bfr[j].bf, (short)0, acc[i][j], false, false);
    }

    #pragma unroll
    for (int i = 0; i < 2; ++i)
        #pragma unroll
        for (int j = 0; j < 4; ++j)
            #pragma unroll
            for (int r = 0; r < 8; ++r) {
                const int m = bm + wm + 16 * i + r + 8 * hi;  // C-tile: row = r + 8*hi
                const int n = bn + wn + 16 * j + lo;          //         col = lane&15
                const float vv = acc[i][j][r];
                if (OUT_BF16) ((u16*)Cout)[(size_t)m * N + n] = f32_to_bf16(vv);
                else          ((float*)Cout)[(size_t)m * N + n] = vv;
            }
}

// ---------------- RoPE on Q, in place, layout (B,S,NH,HD) bf16 ----------------
__global__ void rope_q_kernel(u16* q, const float* __restrict__ cosb,
                              const float* __restrict__ sinb) {
    const int idx = blockIdx.x * blockDim.x + threadIdx.x;
    const int total = B * S * NH * HD2;
    if (idx >= total) return;
    const int d2 = idx % HD2;
    const int h  = (idx / HD2) % NH;
    const int s  = (idx / (HD2 * NH)) % S;
    const int b  = idx / (HD2 * NH * S);
    const size_t base = (((size_t)b * S + s) * NH + h) * HD + 2 * d2;
    const float re = bf16_to_f32(q[base]);
    const float im = bf16_to_f32(q[base + 1]);
    const float c  = cosb[s * HD2 + d2];
    const float sn = sinb[s * HD2 + d2];
    q[base]     = f32_to_bf16(re * c - im * sn);
    q[base + 1] = f32_to_bf16(re * sn + im * c);
}

// ---- RoPE on K + layout change: (B,S,NKV,HD) -> kT (B,NKV,HD,S) bf16 ----
__global__ void rope_k_t_kernel(const u16* __restrict__ kin, const float* __restrict__ cosb,
                                const float* __restrict__ sinb, u16* __restrict__ kT) {
    const int idx = blockIdx.x * blockDim.x + threadIdx.x;
    const int total = B * S * NKV * HD2;
    if (idx >= total) return;
    const int d2 = idx % HD2;
    const int h  = (idx / HD2) % NKV;
    const int s  = (idx / (HD2 * NKV)) % S;
    const int b  = idx / (HD2 * NKV * S);
    const size_t src = (((size_t)b * S + s) * NKV + h) * HD + 2 * d2;
    const float re = bf16_to_f32(kin[src]);
    const float im = bf16_to_f32(kin[src + 1]);
    const float c  = cosb[s * HD2 + d2];
    const float sn = sinb[s * HD2 + d2];
    const size_t dbase = (((size_t)b * NKV + h) * HD);
    kT[(dbase + 2 * d2)     * S + s] = f32_to_bf16(re * c - im * sn);
    kT[(dbase + 2 * d2 + 1) * S + s] = f32_to_bf16(re * sn + im * c);
}

// ---------------- flash attention (causal, GQA) ----------------
// grid.x = B*NH, grid.y = S/128.  8 waves/block, each wave owns 16 q rows.
__global__ __launch_bounds__(256)
void flash_attn_kernel(const u16* __restrict__ q, const u16* __restrict__ kT,
                       const u16* __restrict__ v, u16* __restrict__ o) {
    __shared__ __align__(16) u16 plds[8][16][32];   // per-wave P staging (C->A layout)
    const int tid  = threadIdx.x;
    const int lane = tid & 31;
    const int w    = tid >> 5;
    const int hi   = lane >> 4;
    const int lo   = lane & 15;
    const int bh   = blockIdx.x;
    const int b    = bh / NH;
    const int h    = bh % NH;
    const int kvh  = h / NREP;
    const int q0   = blockIdx.y * 128 + w * 16;

    // Q fragments: 16x128 = 4 chunks of 16x32 (A layout)
    Frag16 qf[4];
    {
        const u16* qbase = q + (((size_t)b * S + q0 + lo) * NH + h) * HD;
        #pragma unroll
        for (int c = 0; c < 4; ++c) {
            const u16* p = qbase + c * 32 + hi * 8;
            qf[c].q[0] = *(const B128*)p;
            qf[c].q[1] = *(const B128*)(p + 16);
        }
    }

    v8f oacc[8] = {};
    float mrow[8], lrow[8];
    #pragma unroll
    for (int r = 0; r < 8; ++r) { mrow[r] = -3.0e38f; lrow[r] = 0.0f; }

    const float scale = 0.088388347648318447f;  // 1/sqrt(128)
    const int qmax = q0 + 15;
    const u16* kThead = kT + ((size_t)b * NKV + kvh) * HD * S;
    const u16* vhead  = v + ((size_t)b * S * NKV + kvh) * HD;

    for (int k0 = 0; k0 <= qmax; k0 += 32) {
        // prefetch next key/value tiles into cache (global_prefetch_b8)
        if (k0 + 32 <= qmax) {
            __builtin_prefetch((const void*)(kThead + (size_t)(lo + 16 * hi) * S + k0 + 32), 0, 0);
            __builtin_prefetch((const void*)(vhead + (size_t)(k0 + 32 + lo + 16 * hi) * NKV * HD), 0, 0);
        }
        // scores: two 16-key tiles, each 4 WMMAs over the 128 head dims
        v8f sacc[2] = {};
        #pragma unroll
        for (int t = 0; t < 2; ++t) {
            #pragma unroll
            for (int c = 0; c < 4; ++c) {
                const int d = c * 32;  // B-frag: lane = key (N), elements = head dims (K)
                const u16* kp = kThead + (size_t)d * S + k0 + 16 * t + lo;
                Frag16 kf;
                #pragma unroll
                for (int e = 0; e < 16; ++e)
                    kf.h[e] = kp[(size_t)((e < 8) ? (hi * 8 + e) : (16 + hi * 8 + e - 8)) * S];
                sacc[t] = __builtin_amdgcn_wmma_f32_16x16x32_bf16(
                    false, qf[c].bf, false, kf.bf, (short)0, sacc[t], false, false);
            }
        }
        // scale + causal mask, online softmax stats
        float p0[8], p1[8], tmax[8];
        #pragma unroll
        for (int r = 0; r < 8; ++r) {
            const int qr = q0 + 8 * hi + r;
            p0[r] = sacc[0][r] * scale + (((k0 + lo)      > qr) ? -1.0e9f : 0.0f);
            p1[r] = sacc[1][r] * scale + (((k0 + 16 + lo) > qr) ? -1.0e9f : 0.0f);
            tmax[r] = fmaxf(p0[r], p1[r]);
        }
        #pragma unroll
        for (int off = 1; off < 16; off <<= 1)
            #pragma unroll
            for (int r = 0; r < 8; ++r)
                tmax[r] = fmaxf(tmax[r], __shfl_xor(tmax[r], off, 32));
        float alpha[8], rs[8];
        #pragma unroll
        for (int r = 0; r < 8; ++r) {
            const float mnew = fmaxf(mrow[r], tmax[r]);
            alpha[r] = __expf(mrow[r] - mnew);
            mrow[r]  = mnew;
            p0[r] = __expf(p0[r] - mnew);
            p1[r] = __expf(p1[r] - mnew);
            rs[r] = p0[r] + p1[r];
        }
        #pragma unroll
        for (int off = 1; off < 16; off <<= 1)
            #pragma unroll
            for (int r = 0; r < 8; ++r)
                rs[r] += __shfl_xor(rs[r], off, 32);
        #pragma unroll
        for (int r = 0; r < 8; ++r) lrow[r] = lrow[r] * alpha[r] + rs[r];
        // rescale O accumulators
        #pragma unroll
        for (int nt = 0; nt < 8; ++nt)
            #pragma unroll
            for (int r = 0; r < 8; ++r)
                oacc[nt][r] *= alpha[r];

        // P: C layout -> A layout via per-wave LDS (DS ops are in-order per wave)
        #pragma unroll
        for (int r = 0; r < 8; ++r) {
            plds[w][8 * hi + r][lo]      = f32_to_bf16(p0[r]);
            plds[w][8 * hi + r][lo + 16] = f32_to_bf16(p1[r]);
        }
        asm volatile("s_wait_dscnt 0" ::: "memory");
        Frag16 pf;
        pf.q[0] = *(const B128*)&plds[w][lo][hi * 8];
        pf.q[1] = *(const B128*)&plds[w][lo][hi * 8 + 16];

        // O += P(16x32) @ V(32x128): 8 WMMAs (one per 16-wide slice of HD)
        const int kk = lo;   // B-frag: lane = hd column (N), elements = keys (K)
        const u16* vrow = vhead + (size_t)k0 * NKV * HD + kk;
        #pragma unroll
        for (int nt = 0; nt < 8; ++nt) {
            Frag16 vf;
            const u16* vp = vrow + nt * 16;
            #pragma unroll
            for (int e = 0; e < 16; ++e)
                vf.h[e] = vp[(size_t)((e < 8) ? (hi * 8 + e) : (16 + hi * 8 + e - 8)) * NKV * HD];
            oacc[nt] = __builtin_amdgcn_wmma_f32_16x16x32_bf16(
                false, pf.bf, false, vf.bf, (short)0, oacc[nt], false, false);
        }
    }

    // epilogue: O /= l, store bf16 (B,S,NH,HD)
    #pragma unroll
    for (int r = 0; r < 8; ++r) lrow[r] = 1.0f / lrow[r];
    #pragma unroll
    for (int nt = 0; nt < 8; ++nt)
        #pragma unroll
        for (int r = 0; r < 8; ++r) {
            const int qr = q0 + 8 * hi + r;
            o[(((size_t)b * S + qr) * NH + h) * HD + nt * 16 + lo] =
                f32_to_bf16(oacc[nt][r] * lrow[r]);
        }
}

// ---------------- host orchestration ----------------
extern "C" void kernel_launch(void* const* d_in, const int* in_sizes, int n_in,
                              void* d_out, int out_size, void* d_ws, size_t ws_size,
                              hipStream_t stream) {
    (void)in_sizes; (void)n_in; (void)out_size; (void)ws_size;
    const float* x  = (const float*)d_in[0];
    const float* wq = (const float*)d_in[1];
    const float* wk = (const float*)d_in[2];
    const float* wv = (const float*)d_in[3];
    const float* wo = (const float*)d_in[4];
    const float* fc = (const float*)d_in[5];
    const float* fs = (const float*)d_in[6];
    // d_in[7] = mask (causal, recomputed on the fly), d_in[8] = start_pos (0)

    char* ws = (char*)d_ws;
    size_t off = 0;
    auto take = [&](size_t bytes) -> char* {
        char* p = ws + off;
        off += (bytes + 255) & ~(size_t)255;
        return p;
    };
    u16* xbf  = (u16*)take((size_t)B * S * D * 2);
    u16* wqT  = (u16*)take((size_t)D * D * 2);
    u16* wkT  = (u16*)take((size_t)D * (NKV * HD) * 2);
    u16* wvT  = (u16*)take((size_t)D * (NKV * HD) * 2);
    u16* woT  = (u16*)take((size_t)D * D * 2);
    u16* qbf  = (u16*)take((size_t)B * S * NH * HD * 2);
    u16* klin = (u16*)take((size_t)B * S * NKV * HD * 2);
    u16* vbf  = (u16*)take((size_t)B * S * NKV * HD * 2);
    u16* kT   = (u16*)take((size_t)B * NKV * HD * S * 2);
    u16* attn = xbf;   // x consumed by QKV GEMMs before flash writes here

    int n = B * S * D;
    cast_bf16_kernel<<<(n + 255) / 256, 256, 0, stream>>>(x, xbf, n);

    dim3 tb(32, 8);
    transpose_cast_kernel<<<dim3(D / 32, (NH * HD) / 32),  tb, 0, stream>>>(wq, wqT, NH * HD,  D);
    transpose_cast_kernel<<<dim3(D / 32, (NKV * HD) / 32), tb, 0, stream>>>(wk, wkT, NKV * HD, D);
    transpose_cast_kernel<<<dim3(D / 32, (NKV * HD) / 32), tb, 0, stream>>>(wv, wvT, NKV * HD, D);
    transpose_cast_kernel<<<dim3((NH * HD) / 32, D / 32),  tb, 0, stream>>>(wo, woT, D, NH * HD);

    const int M = B * S;
    gemm_bf16_kernel<true><<<dim3((NH * HD) / 128, M / 128), 256, 0, stream>>>(
        xbf, wqT, qbf, M, NH * HD, D);
    gemm_bf16_kernel<true><<<dim3((NKV * HD) / 128, M / 128), 256, 0, stream>>>(
        xbf, wkT, klin, M, NKV * HD, D);
    gemm_bf16_kernel<true><<<dim3((NKV * HD) / 128, M / 128), 256, 0, stream>>>(
        xbf, wvT, vbf, M, NKV * HD, D);

    n = B * S * NH * HD2;
    rope_q_kernel<<<(n + 255) / 256, 256, 0, stream>>>(qbf, fc, fs);
    n = B * S * NKV * HD2;
    rope_k_t_kernel<<<(n + 255) / 256, 256, 0, stream>>>(klin, fc, fs, kT);

    flash_attn_kernel<<<dim3(B * NH, S / 128), 256, 0, stream>>>(qbf, kT, vbf, attn);

    gemm_bf16_kernel<false><<<dim3(D / 128, M / 128), 256, 0, stream>>>(
        attn, woT, d_out, M, D, D);
}
